// TrajectoryDiffusion_83356725280815
// MI455X (gfx1250) — matrix operand order repaired and checked by reading, not compiled, not run on hardware
//
#include <hip/hip_runtime.h>
#include <hip/hip_bf16.h>
#include <cstdint>
#include <cstddef>

// ---------------------------------------------------------------------------
// Types / WMMA helpers (CDNA5 / gfx1250, wave32)
// ---------------------------------------------------------------------------
typedef _Float16 half_t;
typedef __attribute__((ext_vector_type(16))) _Float16 v16h;
typedef __attribute__((ext_vector_type(8)))  _Float16 v8h;
typedef __attribute__((ext_vector_type(8)))  float    v8f;

#define ACT_NONE 0
#define ACT_SILU 1
#define ACT_GELU 2

__device__ __forceinline__ v8f wmma16x16x32(v16h a, v16h b, v8f c) {
  // D = A(16x32 f16) * B(32x16 f16) + C(16x16 f32)
  return __builtin_amdgcn_wmma_f32_16x16x32_f16(false, a, false, b, (short)0, c,
                                                false, false);
}

// A-matrix fragment (16x32 f16). ISA 7.12.2: lane L holds row m=L&15;
// element i -> k = 16*(i>>3) + 8*(L>>4) + (i&7)  => two contiguous 16B loads.
__device__ __forceinline__ v16h load_a_frag(const half_t* tile, int ldk, int kc,
                                            int lane) {
  int m = lane & 15, hi = lane >> 4;
  const half_t* p = tile + (size_t)m * ldk + kc + hi * 8;
  union { v16h v; v8h h[2]; } u;
  u.h[0] = *(const v8h*)(p);       // k = kc + 8*hi + [0..7]
  u.h[1] = *(const v8h*)(p + 16);  // k = kc + 16 + 8*hi + [0..7]
  return u.v;
}

// B-matrix fragment (32x16 f16). Weights pre-stored n-major: Wt[n][Kp]
// (k contiguous). Lane L holds col n=n0+(L&15); element i -> k = kc + 16*(i>>3) + 8*(L>>4) + (i&7).
__device__ __forceinline__ v16h load_b_frag(const half_t* __restrict__ Wt, int Kp,
                                            int n0, int kc, int lane) {
  int n = n0 + (lane & 15), hi = lane >> 4;
  const half_t* p = Wt + (size_t)n * Kp + kc + hi * 8;
  union { v16h v; v8h h[2]; } u;
  u.h[0] = *(const v8h*)(p);
  u.h[1] = *(const v8h*)(p + 16);
  return u.v;
}

__device__ __forceinline__ float act_apply(float x, int act) {
  if (act == ACT_SILU) return x / (1.f + __expf(-x));
  if (act == ACT_GELU) return 0.5f * x * (1.f + erff(x * 0.70710678118654752f));
  return x;
}

// ---------------------------------------------------------------------------
// Weight prep: W[K][N] f32  ->  Wt[N][Kp] f16 (zero pad K..Kp)
// ---------------------------------------------------------------------------
__global__ void prep_w_kernel(const float* __restrict__ W, half_t* __restrict__ Wt,
                              int K, int N, int Kp) {
  int i = blockIdx.x * blockDim.x + threadIdx.x;
  if (i >= N * Kp) return;
  int n = i / Kp, kp = i % Kp;
  Wt[i] = (half_t)(kp < K ? W[(size_t)kp * N + n] : 0.f);
}

// ---------------------------------------------------------------------------
// Generic row GEMM: C[M][N] = act(A[M][K] @ W + b) (+resid). K,N mult of 32/16.
// One wave per 16-row tile; A tile staged in LDS as f16.
// ---------------------------------------------------------------------------
__global__ void gemm_rows_kernel(const float* __restrict__ A,
                                 const half_t* __restrict__ Wt,
                                 const float* __restrict__ bias,
                                 float* __restrict__ C,
                                 const float* __restrict__ resid,
                                 int M, int N, int K, int act) {
  extern __shared__ __align__(16) half_t sm[];  // [wavesPerBlock][16][K]
  int wave = threadIdx.x >> 5, lane = threadIdx.x & 31;
  half_t* tile = sm + (size_t)wave * 16 * K;
  int mtiles = (M + 15) >> 4;
  int mt = blockIdx.x * (blockDim.x >> 5) + wave;
  if (mt >= mtiles) return;  // wave-uniform exit; EXEC stays all-ones elsewhere
  int m0 = mt << 4;
  int r = lane & 15, hi = lane >> 4;
  int row = m0 + r;
  int rc = row < M ? row : M - 1;
  const float* a = A + (size_t)rc * K;
  int halfK = K >> 1;
  for (int c = hi * halfK; c < hi * halfK + halfK; ++c)
    tile[(size_t)r * K + c] = (half_t)a[c];
  int nchunks = K >> 5;
  for (int nt = 0; nt < (N >> 4); ++nt) {
    v8f acc = {};
    for (int kc = 0; kc < nchunks; ++kc) {
      v16h af = load_a_frag(tile, K, kc * 32, lane);
      v16h bf = load_b_frag(Wt, K, nt * 16, kc * 32, lane);
      acc = wmma16x16x32(af, bf, acc);
    }
    int n = nt * 16 + (lane & 15);
    float b = bias ? bias[n] : 0.f;
#pragma unroll
    for (int rr = 0; rr < 8; ++rr) {
      int m = m0 + hi * 8 + rr;
      if (m < M) {
        float x = acc[rr] + b;
        if (resid) x += resid[(size_t)m * N + n];
        C[(size_t)m * N + n] = act_apply(x, act);
      }
    }
  }
}

// ---------------------------------------------------------------------------
// Edge-conditioned conv, message stage (fused gather + GEMM + SiLU + scatter):
//   msg = SiLU([h_src | h_dst | ea] @ Wm + bm); atomic mean-accumulate at dst.
// K = 132 padded to 160 (5 chunks of 32), N = 64 (4 tiles). 16 edges per wave.
// ---------------------------------------------------------------------------
#define EM_WAVES 4
__global__ void edge_msg_kernel(const float* __restrict__ h,   // [F][V][64]
                                const int* __restrict__ src,
                                const int* __restrict__ dst,
                                const float* __restrict__ ea,  // [E][4]
                                const half_t* __restrict__ Wm, // [64][160]
                                const float* __restrict__ bm,
                                float* __restrict__ agg,       // [F][V][64]
                                int F, int V, int E) {
  __shared__ __align__(16) half_t sA[EM_WAVES][16][160];
  __shared__ int sDst[EM_WAVES][16];
  int wave = threadIdx.x >> 5, lane = threadIdx.x & 31;
  int tilesPerFrame = (E + 15) >> 4;
  long gtile = (long)blockIdx.x * EM_WAVES + wave;
  long ntiles = (long)F * tilesPerFrame;
  if (gtile >= ntiles) return;  // wave-uniform
  int f = (int)(gtile / tilesPerFrame);
  int e0 = (int)(gtile % tilesPerFrame) << 4;
  int r = lane & 15, hi = lane >> 4;
  int e = e0 + r;
  int ec = e < E ? e : E - 1;
  int s = src[ec], d = dst[ec];
  if (hi == 0) sDst[wave][r] = (e < E) ? d : -1;
  const float* hs = h + ((size_t)f * V + s) * 64;
  const float* hd = h + ((size_t)f * V + d) * 64;
  for (int c = hi * 80; c < hi * 80 + 80; ++c) {
    float val;
    if (c < 64)        val = hs[c];
    else if (c < 128)  val = hd[c - 64];
    else if (c < 132)  val = ea[(size_t)ec * 4 + (c - 128)];
    else               val = 0.f;
    sA[wave][r][c] = (half_t)val;
  }
  v16h afr[5];
#pragma unroll
  for (int kc = 0; kc < 5; ++kc)
    afr[kc] = load_a_frag(&sA[wave][0][0], 160, kc * 32, lane);
  for (int nt = 0; nt < 4; ++nt) {
    v8f acc = {};
#pragma unroll
    for (int kc = 0; kc < 5; ++kc) {
      v16h bf = load_b_frag(Wm, 160, nt * 16, kc * 32, lane);
      acc = wmma16x16x32(afr[kc], bf, acc);
    }
    int n = nt * 16 + (lane & 15);
    float bias = bm[n];
#pragma unroll
    for (int rr = 0; rr < 8; ++rr) {
      int m = hi * 8 + rr;
      int dn = sDst[wave][m];
      if (dn >= 0) {
        float x = acc[rr] + bias;
        float msg = x / (1.f + __expf(-x));  // SiLU
        atomicAdd(&agg[((size_t)f * V + dn) * 64 + n], msg);
      }
    }
  }
}

// ---------------------------------------------------------------------------
// Conv update stage: h = h + ([h | agg/deg] @ Wu + bu).  K=128, N=64, in-place.
// ---------------------------------------------------------------------------
__global__ void gnn_update_kernel(float* __restrict__ h,
                                  const float* __restrict__ agg,
                                  const float* __restrict__ invdeg,
                                  const half_t* __restrict__ Wu,  // [64][128]
                                  const float* __restrict__ bu,
                                  int FV, int V) {
  __shared__ __align__(16) half_t sA[2][16][128];
  int wave = threadIdx.x >> 5, lane = threadIdx.x & 31;
  int mtiles = (FV + 15) >> 4;
  int mt = blockIdx.x * 2 + wave;
  if (mt >= mtiles) return;
  int m0 = mt << 4;
  int r = lane & 15, hi = lane >> 4;
  int row = m0 + r;
  int rc = row < FV ? row : FV - 1;
  float w = invdeg[rc % V];
  const float* hp = h + (size_t)rc * 64;
  const float* ap = agg + (size_t)rc * 64;
  for (int c = hi * 64; c < hi * 64 + 64; ++c)
    sA[wave][r][c] = (half_t)(c < 64 ? hp[c] : ap[c - 64] * w);
  for (int nt = 0; nt < 4; ++nt) {
    v8f acc = {};
#pragma unroll
    for (int kc = 0; kc < 4; ++kc) {
      v16h af = load_a_frag(&sA[wave][0][0], 128, kc * 32, lane);
      v16h bf = load_b_frag(Wu, 128, nt * 16, kc * 32, lane);
      acc = wmma16x16x32(af, bf, acc);
    }
    int n = nt * 16 + (lane & 15);
    float bb = bu[n];
#pragma unroll
    for (int rr = 0; rr < 8; ++rr) {
      int m = m0 + hi * 8 + rr;
      if (m < FV) {
        size_t idx = (size_t)m * 64 + n;
        h[idx] = h[idx] + acc[rr] + bb;  // residual
      }
    }
  }
}

// ---------------------------------------------------------------------------
// Small VALU kernels
// ---------------------------------------------------------------------------
__global__ void deg_kernel(const int* __restrict__ dst, float* __restrict__ deg, int E) {
  int e = blockIdx.x * blockDim.x + threadIdx.x;
  if (e < E) atomicAdd(&deg[dst[e]], 1.f);
}
__global__ void invdeg_kernel(const float* __restrict__ deg, float* __restrict__ inv, int V) {
  int i = blockIdx.x * blockDim.x + threadIdx.x;
  if (i < V) inv[i] = 1.f / fmaxf(deg[i], 1.f);
}
// input linear, K=3 -> N=64
__global__ void lin3_kernel(const float* __restrict__ x, const float* __restrict__ W,
                            const float* __restrict__ b, float* __restrict__ h, long rows) {
  long i = (long)blockIdx.x * blockDim.x + threadIdx.x;
  if (i >= rows * 64) return;
  long row = i >> 6; int n = (int)(i & 63);
  h[i] = x[row * 3 + 0] * W[n] + x[row * 3 + 1] * W[64 + n] +
         x[row * 3 + 2] * W[128 + n] + b[n];
}
// output linear, K=64 -> N=3
__global__ void out3_kernel(const float* __restrict__ h, const float* __restrict__ W,
                            const float* __restrict__ b, float* __restrict__ o, long rows) {
  long i = (long)blockIdx.x * blockDim.x + threadIdx.x;
  if (i >= rows * 3) return;
  long row = i / 3; int n = (int)(i % 3);
  float s = b[n];
  const float* hr = h + row * 64;
#pragma unroll
  for (int k = 0; k < 64; ++k) s += hr[k] * W[k * 3 + n];
  o[i] = s;
}
__global__ void pool_mean_kernel(const float* __restrict__ hw, float* __restrict__ pooled,
                                 int V, int C) {
  int f = blockIdx.x, c = threadIdx.x;
  if (c >= C) return;
  const float* p = hw + (size_t)f * V * C + c;
  float s = 0.f;
  for (int v = 0; v < V; ++v) s += p[(size_t)v * C];
  pooled[(size_t)f * C + c] = s / (float)V;
}
__global__ void time_in_kernel(const float* __restrict__ t, float* __restrict__ out, int Bn) {
  int i = blockIdx.x * blockDim.x + threadIdx.x;
  if (i >= Bn * 64) return;
  int b = i / 64, j = i % 64;
  float e = t[b] * __expf(-0.14619588465f * (float)j);  // -ln(1e4)/63
  out[b * 128 + j] = __sinf(e);
  out[b * 128 + 64 + j] = __cosf(e);
}
__global__ void combine_kernel(float* __restrict__ x, const float* __restrict__ pos,
                               const float* __restrict__ temb, int T, int C, int total) {
  int i = blockIdx.x * blockDim.x + threadIdx.x;
  if (i >= total) return;
  int f = i / C, c = i % C;
  x[i] += pos[(f % T) * C + c] + temb[(f / T) * C + c];
}
__global__ void bcast_kernel(const float* __restrict__ ftmp, float* __restrict__ hA,
                             int V, long total) {
  long i = (long)blockIdx.x * blockDim.x + threadIdx.x;
  if (i >= total) return;
  long fv = i >> 6; int c = (int)(i & 63);
  int f = (int)(fv / V);
  hA[i] = ftmp[f * 64 + c];
}
// attention: T=32, dh=16, one wave per (batch, head)
#define NH_ 8
__global__ void attn_kernel(const float* __restrict__ q, const float* __restrict__ k,
                            const float* __restrict__ v, float* __restrict__ o, int T) {
  int bh = blockIdx.x, b = bh / NH_, h = bh % NH_;
  int lane = threadIdx.x;  // query index
  __shared__ float sk[32][16], sv[32][16];
  size_t base = ((size_t)(b * T + lane)) * 128 + h * 16;
  for (int d = 0; d < 16; ++d) { sk[lane][d] = k[base + d]; sv[lane][d] = v[base + d]; }
  __syncthreads();
  float qr[16];
#pragma unroll
  for (int d = 0; d < 16; ++d) qr[d] = q[base + d];
  float sc[32]; float mx = -1e30f;
#pragma unroll
  for (int tt = 0; tt < 32; ++tt) {
    float s = 0.f;
#pragma unroll
    for (int d = 0; d < 16; ++d) s += qr[d] * sk[tt][d];
    s *= 0.25f;  // 1/sqrt(16)
    sc[tt] = s; mx = fmaxf(mx, s);
  }
  float den = 0.f;
#pragma unroll
  for (int tt = 0; tt < 32; ++tt) { sc[tt] = __expf(sc[tt] - mx); den += sc[tt]; }
  float inv = 1.f / den;
#pragma unroll
  for (int d = 0; d < 16; ++d) {
    float acc = 0.f;
#pragma unroll
    for (int tt = 0; tt < 32; ++tt) acc += sc[tt] * sv[tt][d];
    o[base + d] = acc * inv;
  }
}
// post-norm layernorm: out = LN(a + bres) * g + be ; D == blockDim.x == 128
__global__ void add_ln_kernel(const float* __restrict__ a, const float* __restrict__ bres,
                              const float* __restrict__ g, const float* __restrict__ be,
                              float* __restrict__ out, int D) {
  int row = blockIdx.x, c = threadIdx.x;
  __shared__ float red[128];
  float x = a[(size_t)row * D + c] + bres[(size_t)row * D + c];
  red[c] = x; __syncthreads();
  for (int s = 64; s > 0; s >>= 1) { if (c < s) red[c] += red[c + s]; __syncthreads(); }
  float mu = red[0] / D; __syncthreads();
  float dv = x - mu;
  red[c] = dv * dv; __syncthreads();
  for (int s = 64; s > 0; s >>= 1) { if (c < s) red[c] += red[c + s]; __syncthreads(); }
  float var = red[0] / D;
  out[(size_t)row * D + c] = dv * rsqrtf(var + 1e-5f) * g[c] + be[c];
}

// ---------------------------------------------------------------------------
// Host orchestration
// ---------------------------------------------------------------------------
static void gemm_rows(const float* A, const half_t* Wt, const float* bias, float* C,
                      const float* resid, int M, int N, int K, int act,
                      hipStream_t s) {
  int mtiles = (M + 15) / 16;
  int wavesPerBlock = 2;
  dim3 block(64);
  dim3 grid((mtiles + wavesPerBlock - 1) / wavesPerBlock);
  size_t shmem = (size_t)wavesPerBlock * 16 * K * sizeof(half_t);
  gemm_rows_kernel<<<grid, block, shmem, s>>>(A, Wt, bias, C, resid, M, N, K, act);
}

extern "C" void kernel_launch(void* const* d_in, const int* in_sizes, int n_in,
                              void* d_out, int out_size, void* d_ws, size_t ws_size,
                              hipStream_t stream) {
  const int B = 2, T = 32, V = 2500, E = 15000;
  const int F = B * T;
  const long FV = (long)F * V;
  (void)in_sizes; (void)n_in; (void)out_size; (void)ws_size;

  const float* x   = (const float*)d_in[0];
  const float* tt  = (const float*)d_in[1];
  const int*   ei  = (const int*)d_in[2];
  const int*   src = ei;
  const int*   dst = ei + E;
  const float* ea  = (const float*)d_in[3];
  auto P = [&](int i) { return (const float*)d_in[i]; };

  // Param indices (jax pytree: dict keys sorted; lin_p -> W,b; conv_p -> Wm,Wu,bm,bu)
  // dec:4..23  enc:24..43  pool:44,45  pos:46  tf:47+16l  time:111..114  unpool:115,116
  const int DEC = 4, ENC = 24, POOLW = 44, POOLB = 45, POS = 46, TF = 47;
  const int T1W = 111, T1B = 112, T2W = 113, T2B = 114, UNPW = 115, UNPB = 116;

  // ---- workspace bump allocator ----
  char* wp = (char*)d_ws;
  auto alloc = [&](size_t bytes) -> void* {
    void* p = (void*)wp; wp += (bytes + 255) & ~(size_t)255; return p;
  };
  float* hA     = (float*)alloc(FV * 64 * 4);
  float* agg    = (float*)alloc(FV * 64 * 4);
  float* hwide  = (float*)alloc(FV * 128 * 4);
  float* deg    = (float*)alloc(V * 4);
  float* invdeg = (float*)alloc(V * 4);
  float* pooled = (float*)alloc(F * 128 * 4);
  float* seqX   = (float*)alloc(F * 128 * 4);
  float* seqB   = (float*)alloc(F * 512 * 4);
  float* seqC   = (float*)alloc(F * 128 * 4);
  float* qbuf   = (float*)alloc(F * 128 * 4);
  float* kbuf   = (float*)alloc(F * 128 * 4);
  float* vbuf   = (float*)alloc(F * 128 * 4);
  float* attno  = (float*)alloc(F * 128 * 4);
  float* tin    = (float*)alloc(B * 128 * 4);
  float* tmid   = (float*)alloc(B * 512 * 4);
  float* temb   = (float*)alloc(B * 128 * 4);
  float* unp    = (float*)alloc(F * 128 * 4);
  float* ftmp   = (float*)alloc(F * 64 * 4);

  half_t* encWm[4]; half_t* encWu[4]; half_t* decWm[4]; half_t* decWu[4];
  for (int l = 0; l < 4; ++l) {
    encWm[l] = (half_t*)alloc(64 * 160 * 2);
    encWu[l] = (half_t*)alloc(64 * 128 * 2);
    decWm[l] = (half_t*)alloc(64 * 160 * 2);
    decWu[l] = (half_t*)alloc(64 * 128 * 2);
  }
  half_t* encOutW = (half_t*)alloc(128 * 64 * 2);
  half_t* decInpW = (half_t*)alloc(64 * 128 * 2);
  half_t* poolW   = (half_t*)alloc(128 * 128 * 2);
  half_t* unpoolW = (half_t*)alloc(128 * 128 * 2);
  half_t* t1W     = (half_t*)alloc(512 * 128 * 2);
  half_t* t2W     = (half_t*)alloc(128 * 512 * 2);
  half_t *tfQ[4], *tfK[4], *tfV[4], *tfO[4], *tfF1[4], *tfF2[4];
  for (int l = 0; l < 4; ++l) {
    tfQ[l]  = (half_t*)alloc(128 * 128 * 2);
    tfK[l]  = (half_t*)alloc(128 * 128 * 2);
    tfV[l]  = (half_t*)alloc(128 * 128 * 2);
    tfO[l]  = (half_t*)alloc(128 * 128 * 2);
    tfF1[l] = (half_t*)alloc(512 * 128 * 2);
    tfF2[l] = (half_t*)alloc(128 * 512 * 2);
  }

  auto prep = [&](const float* W, half_t* Wt, int K, int N, int Kp) {
    int n = N * Kp;
    prep_w_kernel<<<(n + 255) / 256, 256, 0, stream>>>(W, Wt, K, N, Kp);
  };

  // ---- weight prep (f16, n-major, K padded to mult of 32) ----
  for (int l = 0; l < 4; ++l) {
    prep(P(ENC + 2 + 4 * l), encWm[l], 132, 64, 160);
    prep(P(ENC + 3 + 4 * l), encWu[l], 128, 64, 128);
    prep(P(DEC + 2 + 4 * l), decWm[l], 132, 64, 160);
    prep(P(DEC + 3 + 4 * l), decWu[l], 128, 64, 128);
  }
  prep(P(ENC + 18), encOutW, 64, 128, 64);
  prep(P(DEC + 0),  decInpW, 128, 64, 128);
  prep(P(POOLW), poolW, 128, 128, 128);
  prep(P(UNPW),  unpoolW, 128, 128, 128);
  prep(P(T1W), t1W, 128, 512, 128);
  prep(P(T2W), t2W, 512, 128, 512);
  for (int l = 0; l < 4; ++l) {
    int b = TF + 16 * l;
    prep(P(b + 0),  tfF1[l], 128, 512, 128);
    prep(P(b + 2),  tfF2[l], 512, 128, 512);
    prep(P(b + 4),  tfK[l], 128, 128, 128);
    prep(P(b + 10), tfO[l], 128, 128, 128);
    prep(P(b + 12), tfQ[l], 128, 128, 128);
    prep(P(b + 14), tfV[l], 128, 128, 128);
  }

  // ---- degrees ----
  hipMemsetAsync(deg, 0, V * 4, stream);
  deg_kernel<<<(E + 255) / 256, 256, 0, stream>>>(dst, deg, E);
  invdeg_kernel<<<(V + 255) / 256, 256, 0, stream>>>(deg, invdeg, V);

  // ---- encoder GNN ----
  lin3_kernel<<<(int)((FV * 64 + 255) / 256), 256, 0, stream>>>(x, P(ENC), P(ENC + 1), hA, FV);
  int tilesPF = (E + 15) / 16;
  long emWaves = (long)F * tilesPF;
  int emBlocks = (int)((emWaves + EM_WAVES - 1) / EM_WAVES);
  int updBlocks = (int)(((FV + 15) / 16 + 1) / 2);
  for (int l = 0; l < 4; ++l) {
    hipMemsetAsync(agg, 0, FV * 64 * 4, stream);
    edge_msg_kernel<<<emBlocks, 32 * EM_WAVES, 0, stream>>>(
        hA, src, dst, ea, encWm[l], P(ENC + 4 + 4 * l), agg, F, V, E);
    gnn_update_kernel<<<updBlocks, 64, 0, stream>>>(
        hA, agg, invdeg, encWu[l], P(ENC + 5 + 4 * l), (int)FV, V);
  }
  gemm_rows(hA, encOutW, P(ENC + 19), hwide, nullptr, (int)FV, 128, 64, ACT_NONE, stream);

  // ---- pool + embeddings ----
  pool_mean_kernel<<<F, 128, 0, stream>>>(hwide, pooled, V, 128);
  gemm_rows(pooled, poolW, P(POOLB), seqX, nullptr, F, 128, 128, ACT_SILU, stream);
  time_in_kernel<<<1, 128, 0, stream>>>(tt, tin, B);
  gemm_rows(tin, t1W, P(T1B), tmid, nullptr, B, 512, 128, ACT_SILU, stream);
  gemm_rows(tmid, t2W, P(T2B), temb, nullptr, B, 128, 512, ACT_NONE, stream);
  combine_kernel<<<(F * 128 + 255) / 256, 256, 0, stream>>>(seqX, P(POS), temb, T, 128, F * 128);

  // ---- transformer (4 post-norm layers) ----
  for (int l = 0; l < 4; ++l) {
    int b = TF + 16 * l;
    gemm_rows(seqX, tfQ[l], P(b + 13), qbuf, nullptr, F, 128, 128, ACT_NONE, stream);
    gemm_rows(seqX, tfK[l], P(b + 5),  kbuf, nullptr, F, 128, 128, ACT_NONE, stream);
    gemm_rows(seqX, tfV[l], P(b + 15), vbuf, nullptr, F, 128, 128, ACT_NONE, stream);
    attn_kernel<<<B * NH_, 32, 0, stream>>>(qbuf, kbuf, vbuf, attno, T);
    gemm_rows(attno, tfO[l], P(b + 11), seqC, nullptr, F, 128, 128, ACT_NONE, stream);
    add_ln_kernel<<<F, 128, 0, stream>>>(seqX, seqC, P(b + 7), P(b + 6), seqX, 128);
    gemm_rows(seqX, tfF1[l], P(b + 1), seqB, nullptr, F, 512, 128, ACT_GELU, stream);
    gemm_rows(seqB, tfF2[l], P(b + 3), seqC, nullptr, F, 128, 512, ACT_NONE, stream);
    add_ln_kernel<<<F, 128, 0, stream>>>(seqX, seqC, P(b + 9), P(b + 8), seqX, 128);
  }

  // ---- unpool -> decoder ----
  gemm_rows(seqX, unpoolW, P(UNPB), unp, nullptr, F, 128, 128, ACT_SILU, stream);
  // dec input linear is identical for all V nodes in a frame: compute per-frame, broadcast
  gemm_rows(unp, decInpW, P(DEC + 1), ftmp, nullptr, F, 64, 128, ACT_NONE, stream);
  bcast_kernel<<<(int)((FV * 64 + 255) / 256), 256, 0, stream>>>(ftmp, hA, V, FV * 64);
  for (int l = 0; l < 4; ++l) {
    hipMemsetAsync(agg, 0, FV * 64 * 4, stream);
    edge_msg_kernel<<<emBlocks, 32 * EM_WAVES, 0, stream>>>(
        hA, src, dst, ea, decWm[l], P(DEC + 4 + 4 * l), agg, F, V, E);
    gnn_update_kernel<<<updBlocks, 64, 0, stream>>>(
        hA, agg, invdeg, decWu[l], P(DEC + 5 + 4 * l), (int)FV, V);
  }
  out3_kernel<<<(int)((FV * 3 + 255) / 256), 256, 0, stream>>>(
      hA, P(DEC + 18), P(DEC + 19), (float*)d_out, FV);
}
